// QuantLlmLinear_48550310314235
// MI455X (gfx1250) — matrix operand order repaired and checked by reading, not compiled, not run on hardware
//
#include <hip/hip_runtime.h>

// FP6-E3M2 dequant + GEMM + bias for gfx1250 (MI455X).
// Memory-bound on the int32 code stream (235 MB @ 23.3 TB/s ~ 10.5 us floor);
// compute via v_wmma_f32_16x16x32_f16.

typedef __attribute__((ext_vector_type(16))) _Float16 v16h;
typedef __attribute__((ext_vector_type(8)))  _Float16 v8h;
typedef __attribute__((ext_vector_type(8)))  float    v8f;

#define BM 128
#define BN 128
#define BK 32
#define LDA 40   // padded LDS row stride in halfs (80B, 16B-aligned rows)
#define LDB 40

// FP6 E3M2 code -> fp16 bit pattern equal to (true value) * 2^-12.
// sign: bit5 -> bit15 ; exp(3b)+man(2b): bits4:0 -> fp16 bits 12:8.
__device__ __forceinline__ unsigned int fp6_pair(int c0, int c1) {
    unsigned int h0 = ((unsigned)(c0 & 0x20) << 10) | ((unsigned)(c0 & 0x1F) << 8);
    unsigned int h1 = ((unsigned)(c1 & 0x20) << 10) | ((unsigned)(c1 & 0x1F) << 8);
    return h0 | (h1 << 16);
}

__global__ __launch_bounds__(256) void fp6_gemm_wmma(
    const float* __restrict__ x,        // [M,K] f32
    const int*   __restrict__ w,        // [N,K] int32 (FP6 codes)
    const float* __restrict__ scales,   // [N]
    const float* __restrict__ bias,     // [N]
    float*       __restrict__ out,      // [M,N]
    int M, int K, int N)
{
    __shared__ __align__(16) _Float16 As[BM * LDA];
    __shared__ __align__(16) _Float16 Bs[BN * LDB];

    const int tid  = threadIdx.x;
    const int lane = tid & 31;
    const int wave = tid >> 5;
    const int g    = lane >> 4;   // lane group (0: lanes 0-15, 1: lanes 16-31)
    const int ln   = lane & 15;

    const int m0 = blockIdx.x * BM;   // M fastest-varying -> L2 reuse of w panels
    const int n0 = blockIdx.y * BN;

    const int m_wave = (wave & 3) * 32;   // 4 waves along M
    const int n_wave = (wave >> 2) * 64;  // 2 waves along N

    v8f acc[2][4];
#pragma unroll
    for (int mi = 0; mi < 2; ++mi)
#pragma unroll
        for (int ni = 0; ni < 4; ++ni)
            acc[mi][ni] = (v8f){0.f, 0.f, 0.f, 0.f, 0.f, 0.f, 0.f, 0.f};

    // Cooperative loaders: 2 threads per row, 16 elements each.
    const int lrow = tid >> 1;          // 0..127
    const int lcol = (tid & 1) * 16;    // 0 or 16

    const float* xg = x + (size_t)(m0 + lrow) * K + lcol;
    const int*   wg = w + (size_t)(n0 + lrow) * K + lcol;

    for (int k0 = 0; k0 < K; k0 += BK) {
        // ---- global loads (vectorized 16B) ----
        float4 xa[4];
        int4   wa[4];
#pragma unroll
        for (int q = 0; q < 4; ++q) {
            xa[q] = ((const float4*)(xg + k0))[q];
            wa[q] = ((const int4*)(wg + k0))[q];
        }
        if (k0 + BK < K) {  // prefetch next K panel -> global_prefetch_b8
            __builtin_prefetch(xg + k0 + BK, 0, 3);
            __builtin_prefetch(wg + k0 + BK, 0, 3);
        }

        __syncthreads();  // previous iteration's LDS reads done

        // ---- A: f32 -> f16, store [m][k] ----
        {
            v8h h0, h1;
#pragma unroll
            for (int j = 0; j < 4; ++j) {
                h0[j]     = (_Float16)(&xa[0].x)[j];
                h0[4 + j] = (_Float16)(&xa[1].x)[j];
                h1[j]     = (_Float16)(&xa[2].x)[j];
                h1[4 + j] = (_Float16)(&xa[3].x)[j];
            }
            *(v8h*)&As[lrow * LDA + lcol]     = h0;
            *(v8h*)&As[lrow * LDA + lcol + 8] = h1;
        }
        // ---- B: FP6 code -> fp16*2^-12, store [n][k] ----
        {
            unsigned int* bdst = (unsigned int*)&Bs[lrow * LDB + lcol];
#pragma unroll
            for (int q = 0; q < 4; ++q) {
                bdst[2 * q]     = fp6_pair((&wa[q].x)[0], (&wa[q].x)[1]);
                bdst[2 * q + 1] = fp6_pair((&wa[q].x)[2], (&wa[q].x)[3]);
            }
        }
        __syncthreads();

        // ---- fragment loads (documented gfx1250 WMMA VGPR layouts) ----
        v16h af[2], bf[4];
#pragma unroll
        for (int mi = 0; mi < 2; ++mi) {
            const _Float16* arow = &As[(m_wave + mi * 16 + ln) * LDA];
            // lane group g: halfs 0..7 = K[g*8 .. g*8+7], halfs 8..15 = K[16+g*8 ..]
            v8h lo = *(const v8h*)(arow + g * 8);
            v8h hi = *(const v8h*)(arow + 16 + g * 8);
            af[mi] = __builtin_shufflevector(lo, hi, 0, 1, 2, 3, 4, 5, 6, 7,
                                             8, 9, 10, 11, 12, 13, 14, 15);
        }
#pragma unroll
        for (int ni = 0; ni < 4; ++ni) {
            // B col = lane&15; lane group g: halfs 0..7 = K[g*8 .. g*8+7], halfs 8..15 = K[16+g*8 ..]
            const _Float16* brow = &Bs[(n_wave + ni * 16 + ln) * LDB + g * 8];
            v8h lo = *(const v8h*)(brow);
            v8h hi = *(const v8h*)(brow + 16);
            bf[ni] = __builtin_shufflevector(lo, hi, 0, 1, 2, 3, 4, 5, 6, 7,
                                             8, 9, 10, 11, 12, 13, 14, 15);
        }

        // ---- 8x v_wmma_f32_16x16x32_f16 per wave per K step ----
#pragma unroll
        for (int mi = 0; mi < 2; ++mi)
#pragma unroll
            for (int ni = 0; ni < 4; ++ni)
                acc[mi][ni] = __builtin_amdgcn_wmma_f32_16x16x32_f16(
                    false, af[mi], false, bf[ni],
                    (short)0, acc[mi][ni], false, false);
    }

    // ---- epilogue: out = acc * 4096 * scale[n] + bias[n] ----
#pragma unroll
    for (int ni = 0; ni < 4; ++ni) {
        const int n = n0 + n_wave + ni * 16 + ln;
        const float s  = scales[n] * 4096.0f;
        const float bb = bias[n];
#pragma unroll
        for (int mi = 0; mi < 2; ++mi) {
            const int mbase = m0 + m_wave + mi * 16 + g * 8;  // C/D layout: VGPR v -> M = g*8+v
#pragma unroll
            for (int v = 0; v < 8; ++v)
                out[(size_t)(mbase + v) * N + n] = acc[mi][ni][v] * s + bb;
        }
    }
}

extern "C" void kernel_launch(void* const* d_in, const int* in_sizes, int n_in,
                              void* d_out, int out_size, void* d_ws, size_t ws_size,
                              hipStream_t stream) {
    const float* x      = (const float*)d_in[0];
    const int*   w      = (const int*)d_in[1];
    const float* scales = (const float*)d_in[2];
    const float* bias   = (const float*)d_in[3];
    float*       out    = (float*)d_out;

    const int N = in_sizes[2];
    const int K = in_sizes[1] / N;
    const int M = in_sizes[0] / K;

    dim3 grid(M / BM, N / BN);   // M=256,K=4096,N=14336 all divide tile sizes exactly
    fp6_gemm_wmma<<<grid, 256, 0, stream>>>(x, w, scales, bias, out, M, K, N);
}
